// ContrastiveLoss_65266323030067
// MI455X (gfx1250) — compile-verified, matrix operand order checked by reading, not run on hardware
//
#include <hip/hip_runtime.h>
#include <hip/hip_bf16.h>
#include <math.h>

typedef __attribute__((ext_vector_type(16))) __bf16 v16bf;
typedef __attribute__((ext_vector_type(8)))  __bf16 v8bf;
typedef __attribute__((ext_vector_type(8)))  float  v8f;
typedef __attribute__((ext_vector_type(4)))  unsigned int v4u;
typedef __attribute__((ext_vector_type(8)))  int v8i;
typedef __attribute__((ext_vector_type(4)))  int v4i;

#define NROWS 8192
#define DOUT  256
#define LDS_STRIDE 40   // 32 bf16 payload + 8 pad -> 80B rows (TDM pad: 16dw + 4dw)

// ---- fragment builders -----------------------------------------------------
// A/B fragment for v_wmma_f32_16x16x32_bf16 (wave32):
// lane l: p = l&15 (row of A / col of B^T), h = l>>4.
// vector elems 0..7  = 8 contiguous bf16 at K = k0 + h*8
// vector elems 8..15 = 8 contiguous bf16 at K = k0 + 16 + h*8

static __device__ __forceinline__ v16bf frag_from_f32(const float* __restrict__ base) {
  v16bf f;
#pragma unroll
  for (int i = 0; i < 8; ++i) {
    f[i]     = (__bf16)base[i];
    f[i + 8] = (__bf16)base[i + 16];
  }
  return f;
}

// Reads 8 bf16 at base[0..7] and 8 at base[16..23] (works for global or LDS).
static __device__ __forceinline__ v16bf frag_from_bf16(const __bf16* base) {
  const v8bf lo = *reinterpret_cast<const v8bf*>(base);
  const v8bf hi = *reinterpret_cast<const v8bf*>(base + 16);
  v16bf f;
#pragma unroll
  for (int i = 0; i < 8; ++i) { f[i] = lo[i]; f[i + 8] = hi[i]; }
  return f;
}

// ---- TDM helpers -----------------------------------------------------------
// D# group0: {count=1, lds_addr, global_addr[31:0], global_addr[56:32]|type=2}
static __device__ __forceinline__ v4u tdm_g0(unsigned int lds_off, unsigned long long ga) {
  v4u g0;
  g0[0] = 1u;                                   // count=1 valid descriptor
  g0[1] = lds_off;                              // LDS byte address of tile start
  g0[2] = (unsigned int)ga;                     // global_addr lo
  g0[3] = (unsigned int)(ga >> 32) | (2u << 30);// global_addr hi | type=2 (image)
  return g0;
}

static __device__ __forceinline__ void tdm_load(v4u g0, v8i g1) {
  v4i z = {};
#if __clang_major__ >= 23
  v8i z8 = {};
  __builtin_amdgcn_tensor_load_to_lds(g0, g1, z, z, z8, 0);
#else
  __builtin_amdgcn_tensor_load_to_lds(g0, g1, z, z, 0);
#endif
}

// ---- kernel 0: zero the reduction buffers ----------------------------------
__global__ void zero_kernel(float* __restrict__ p, int n) {
  int i = blockIdx.x * blockDim.x + threadIdx.x;
  if (i < n) p[i] = 0.0f;
}

// ---- kernel 1: P = A @ W^T + b, then row-L2-normalize, store bf16 ----------
__global__ __launch_bounds__(256) void proj_norm_kernel(
    const float* __restrict__ A,     // [M, K] row-major
    const float* __restrict__ W,     // [256, K] row-major (W^T cols = W rows)
    const float* __restrict__ bias,  // [256]
    int K,
    __bf16* __restrict__ Pn) {       // [M, 256] bf16, L2-normalized rows
  __shared__ float tile[16][DOUT + 1];
  __shared__ float norm2[16];

  const int tid = threadIdx.x;
  const int w   = tid >> 5;
  const int l   = tid & 31;
  const int n0  = l & 15;
  const int h   = l >> 4;
  const int rowBase = blockIdx.x * 16;

  v8f acc0, acc1;
#pragma unroll
  for (int r = 0; r < 8; ++r) { acc0[r] = 0.0f; acc1[r] = 0.0f; }

  const float* arow  = A + (size_t)(rowBase + n0) * K + h * 8;
  const float* wrow0 = W + (size_t)(w * 32 + n0) * K + h * 8;
  const float* wrow1 = W + (size_t)(w * 32 + 16 + n0) * K + h * 8;

  for (int k0 = 0; k0 < K; k0 += 32) {
    if (k0 + 256 < K) {                         // uniform; global_prefetch_b8
      __builtin_prefetch(arow + k0 + 256, 0, 0);
      __builtin_prefetch(wrow0 + k0 + 256, 0, 0);
      __builtin_prefetch(wrow1 + k0 + 256, 0, 0);
    }
    v16bf a  = frag_from_f32(arow  + k0);
    v16bf b0 = frag_from_f32(wrow0 + k0);
    v16bf b1 = frag_from_f32(wrow1 + k0);
    acc0 = __builtin_amdgcn_wmma_f32_16x16x32_bf16(false, a, false, b0, (short)0, acc0, false, false);
    acc1 = __builtin_amdgcn_wmma_f32_16x16x32_bf16(false, a, false, b1, (short)0, acc1, false, false);
  }

  const float b0v = bias[w * 32 + n0];
  const float b1v = bias[w * 32 + 16 + n0];
#pragma unroll
  for (int r = 0; r < 8; ++r) {
    const int m = r + 8 * h;                 // C/D layout: m = r + 8*(lane/16)
    tile[m][w * 32 + n0]      = acc0[r] + b0v;
    tile[m][w * 32 + 16 + n0] = acc1[r] + b1v;
  }
  if (tid < 16) norm2[tid] = 0.0f;
  __syncthreads();
  {
    const int row = tid >> 4, seg = tid & 15;
    float p = 0.0f;
#pragma unroll
    for (int c = 0; c < 16; ++c) { float v = tile[row][seg * 16 + c]; p += v * v; }
    atomicAdd(&norm2[row], p);
  }
  __syncthreads();
  {
    const int row = tid >> 4, seg = tid & 15;
    const float inv = 1.0f / fmaxf(sqrtf(norm2[row]), 1e-8f);
    __bf16* dst = Pn + (size_t)(rowBase + row) * DOUT;
#pragma unroll
    for (int c = 0; c < 16; ++c) {
      const int col = seg * 16 + c;
      dst[col] = (__bf16)(tile[row][col] * inv);
    }
  }
}

// ---- kernel 2: fused sim = Xn @ Yn^T, exp, row/col sums, diagonal ----------
// Block = 256 threads (8 waves) computes a 128x128 tile. Waves are a 4x2
// grid: wave (wr, wc) owns rows [wr*32,+32) x cols [wc*64,+64) => 2 A-frags,
// 4 B-frags, 8 accumulators. Per k-step the 128x32 X/Y slices are moved
// into double-buffered LDS by the Tensor Data Mover (one tensor_load_to_lds
// per matrix, hardware row-padding 64B+16B = 80B stride), overlapped with
// WMMA compute and synchronized via TENSORcnt + one barrier per k-step.
__global__ __launch_bounds__(256) void sim_kernel(
    const __bf16* __restrict__ Xn,   // [8192, 256]
    const __bf16* __restrict__ Yn,   // [8192, 256]
    float* __restrict__ rowsum,      // [8192] sum_j exp(sim[i,j])
    float* __restrict__ colsum,      // [8192] sum_i exp(sim[i,j])
    float* __restrict__ diagv) {     // [8192] exp(sim[j,j])
  __shared__ __bf16 Xs[2][128 * LDS_STRIDE];
  __shared__ __bf16 Ys[2][128 * LDS_STRIDE];

  const int tid = threadIdx.x;
  const int w   = tid >> 5;
  const int l   = tid & 31;
  const int n0  = l & 15;
  const int h   = l >> 4;
  const int wr  = w >> 1;           // 0..3 : row group
  const int wc  = w & 1;            // 0..1 : col group
  const int blockRow = blockIdx.y * 128;
  const int colBase  = blockIdx.x * 128;

  // D# group1 (identical for X and Y slices):
  // data_size=2 (4B), pad_enable, pad_interval=3 (16 dwords), pad_amount=3 (4 dwords)
  // tensor_dim0=128 dwords, tensor_dim1=8192 rows, tile=16x128 dwords,
  // tensor_dim0_stride=128 dwords (512B row pitch).
  v8i g1;
  g1[0] = (int)((2u << 16) | (1u << 20) | (3u << 22) | (3u << 25));
  g1[1] = (int)(128u << 16);        // tensor_dim0[15:0]
  g1[2] = (int)(8192u << 16);       // tensor_dim0[31:16]=0 | tensor_dim1[15:0]
  g1[3] = (int)(16u << 16);         // tensor_dim1[31:16]=0 | tile_dim0=16
  g1[4] = 128;                      // tile_dim1=128 | tile_dim2=0
  g1[5] = 128;                      // tensor_dim0_stride[31:0]
  g1[6] = 0;
  g1[7] = 0;

  const unsigned int ldsX[2] = { (unsigned int)(size_t)&Xs[0][0],
                                 (unsigned int)(size_t)&Xs[1][0] };
  const unsigned int ldsY[2] = { (unsigned int)(size_t)&Ys[0][0],
                                 (unsigned int)(size_t)&Ys[1][0] };
  const unsigned long long gaX =
      (unsigned long long)(size_t)Xn + (unsigned long long)blockRow * (DOUT * 2);
  const unsigned long long gaY =
      (unsigned long long)(size_t)Yn + (unsigned long long)colBase * (DOUT * 2);

  v8f acc[2][4];
#pragma unroll
  for (int a = 0; a < 2; ++a)
#pragma unroll
    for (int t = 0; t < 4; ++t)
#pragma unroll
      for (int r = 0; r < 8; ++r) acc[a][t][r] = 0.0f;

  const int arow0 = wr * 32 + n0;
  const int arow1 = arow0 + 16;
  const int NK = DOUT / 32;         // 8 k-steps (64B of K per step)

  if (w == 0) {                     // kick off k-slice 0 into buffer 0
    tdm_load(tdm_g0(ldsX[0], gaX), g1);
    tdm_load(tdm_g0(ldsY[0], gaY), g1);
  }

  for (int kk = 0; kk < NK; ++kk) {
    if (w == 0) __builtin_amdgcn_s_wait_tensorcnt(0);  // current buffer landed
    __syncthreads();                // publish LDS; also: all prior reads done
    if (w == 0 && kk + 1 < NK) {    // DMA next slice while computing this one
      const unsigned long long o = (unsigned long long)(kk + 1) * 64;
      tdm_load(tdm_g0(ldsX[(kk + 1) & 1], gaX + o), g1);
      tdm_load(tdm_g0(ldsY[(kk + 1) & 1], gaY + o), g1);
    }

    const __bf16* xb = Xs[kk & 1];
    const __bf16* yb = Ys[kk & 1];
    const v16bf af0 = frag_from_bf16(&xb[arow0 * LDS_STRIDE + h * 8]);
    const v16bf af1 = frag_from_bf16(&xb[arow1 * LDS_STRIDE + h * 8]);
#pragma unroll
    for (int t = 0; t < 4; ++t) {
      const v16bf bf = frag_from_bf16(&yb[(wc * 64 + t * 16 + n0) * LDS_STRIDE + h * 8]);
      acc[0][t] = __builtin_amdgcn_wmma_f32_16x16x32_bf16(false, af0, false, bf, (short)0, acc[0][t], false, false);
      acc[1][t] = __builtin_amdgcn_wmma_f32_16x16x32_bf16(false, af1, false, bf, (short)0, acc[1][t], false, false);
    }
  }

  // e = exp(sim / tau), tau == 1 (in place)
#pragma unroll
  for (int a = 0; a < 2; ++a)
#pragma unroll
    for (int t = 0; t < 4; ++t)
#pragma unroll
      for (int r = 0; r < 8; ++r) acc[a][t][r] = __expf(acc[a][t][r]);

  // Row sums over this block's 128 columns for row i = blockRow+wr*32+a*16+r+8h.
#pragma unroll
  for (int a = 0; a < 2; ++a) {
#pragma unroll
    for (int r = 0; r < 8; ++r) {
      float s = acc[a][0][r] + acc[a][1][r] + acc[a][2][r] + acc[a][3][r];
#pragma unroll
      for (int off = 1; off < 16; off <<= 1) s += __shfl_xor(s, off, 32);
      if (n0 == 0) atomicAdd(&rowsum[blockRow + wr * 32 + a * 16 + r + 8 * h], s);
    }
  }

  // Col sums (+ diagonal): sum this wave's 32 rows for col j, fold lane-halves.
#pragma unroll
  for (int t = 0; t < 4; ++t) {
    const int j = colBase + wc * 64 + t * 16 + n0;
    float c = 0.0f;
#pragma unroll
    for (int a = 0; a < 2; ++a) {
#pragma unroll
      for (int r = 0; r < 8; ++r) {
        const int i = blockRow + wr * 32 + a * 16 + r + 8 * h;
        if (i == j) diagv[i] = acc[a][t][r];
        c += acc[a][t][r];
      }
    }
    c += __shfl_xor(c, 16, 32);
    if (h == 0) atomicAdd(&colsum[j], c);
  }
}

// ---- kernel 3: epilogue ----------------------------------------------------
// out[j] = log(colsum-pos) + log(rowsum-pos) - 2*log(pos)
__global__ void final_kernel(const float* __restrict__ rowsum,
                             const float* __restrict__ colsum,
                             const float* __restrict__ diagv,
                             float* __restrict__ out) {
  const int i = blockIdx.x * blockDim.x + threadIdx.x;
  if (i < NROWS) {
    const float pos = diagv[i];
    out[i] = logf(colsum[i] - pos) + logf(rowsum[i] - pos) - 2.0f * logf(pos);
  }
}

extern "C" void kernel_launch(void* const* d_in, const int* in_sizes, int n_in,
                              void* d_out, int out_size, void* d_ws, size_t ws_size,
                              hipStream_t stream) {
  (void)in_sizes; (void)n_in; (void)out_size; (void)ws_size;
  const float* X  = (const float*)d_in[0];
  const float* Y  = (const float*)d_in[1];
  const float* Wx = (const float*)d_in[2];
  const float* bx = (const float*)d_in[3];
  const float* Wy = (const float*)d_in[4];
  const float* by = (const float*)d_in[5];
  float* out = (float*)d_out;

  char* ws = (char*)d_ws;
  const size_t nbPn = (size_t)NROWS * DOUT * sizeof(__bf16);   // 4 MB each
  __bf16* Xn     = (__bf16*)ws;
  __bf16* Yn     = (__bf16*)(ws + nbPn);
  float*  rowsum = (float*)(ws + 2 * nbPn);
  float*  colsum = rowsum + NROWS;
  float*  diagv  = colsum + NROWS;

  zero_kernel<<<(2 * NROWS + 255) / 256, 256, 0, stream>>>(rowsum, 2 * NROWS);
  proj_norm_kernel<<<NROWS / 16, 256, 0, stream>>>(X, Wx, bx, 1024, Xn);
  proj_norm_kernel<<<NROWS / 16, 256, 0, stream>>>(Y, Wy, by,  768, Yn);
  dim3 simGrid(NROWS / 128, NROWS / 128);
  sim_kernel<<<simGrid, 256, 0, stream>>>(Xn, Yn, rowsum, colsum, diagv);
  final_kernel<<<NROWS / 256, 256, 0, stream>>>(rowsum, colsum, diagv, out);
}
